// GNN_NET_1305670058226
// MI455X (gfx1250) — matrix operand order repaired and checked by reading, not compiled, not run on hardware
//
#include <hip/hip_runtime.h>

// CDNA5 / gfx1250 WMMA types
typedef __attribute__((ext_vector_type(16))) __bf16 v16bf;
typedef __attribute__((ext_vector_type(8)))  float  v8f;

#define N_NODES 100000
#define N_EDGES 600000
#define N_LABEL 200000
// D_IN = D_HID = 128, D_OUT = 64

// ---------------------------------------------------------------------------
// Degree count: one thread per edge, float atomic on destination node.
// ---------------------------------------------------------------------------
__global__ void count_edges(const int* __restrict__ dst, float* __restrict__ cnt, int E) {
    int e = blockIdx.x * blockDim.x + threadIdx.x;
    if (e < E) atomicAdd(&cnt[dst[e]], 1.0f);
}

__global__ void inv_counts(float* __restrict__ cnt, int N) {
    int i = blockIdx.x * blockDim.x + threadIdx.x;
    if (i < N) cnt[i] = 1.0f / fmaxf(cnt[i], 1.0f);
}

// ---------------------------------------------------------------------------
// Scatter-add of 128-float feature rows: one wave32 per edge, 4 floats/lane.
// 512B coalesced read of x[src], 4x global_atomic_add_f32 into agg[dst].
// Node arrays (~130MB) fit the 192MB L2, so these atomics resolve in L2.
// ---------------------------------------------------------------------------
__global__ void scatter_add_feats(const float* __restrict__ feat,
                                  const int* __restrict__ src,
                                  const int* __restrict__ dst,
                                  float* __restrict__ agg, int E) {
    int t    = blockIdx.x * blockDim.x + threadIdx.x;
    int e    = t >> 5;
    int lane = t & 31;
    if (e >= E) return;
    int s = src[e];
    int d = dst[e];
    float4 v = *reinterpret_cast<const float4*>(feat + (size_t)s * 128 + lane * 4);
    float* o = agg + (size_t)d * 128 + lane * 4;
    atomicAdd(o + 0, v.x);
    atomicAdd(o + 1, v.y);
    atomicAdd(o + 2, v.z);
    atomicAdd(o + 3, v.w);
}

// ---------------------------------------------------------------------------
// Fused SAGE linear:  out = [relu]( (agg*inv) @ Wl^T + bias + self @ Wr^T )
//
// Block = NCT waves; wave w owns output-column tile ct=w and keeps ALL of its
// weight fragments (both matrices, all 4 k-steps = 128 bf16/lane) resident in
// VGPRs for the whole kernel. The block walks RTILES row tiles: the 16x128
// agg rows (scaled by 1/deg) and self rows are staged once per row tile into
// LDS as bf16 by all threads; every wave then issues all 16 ds_load_b128 for
// its A fragments before the WMMA burst so loads overlap the matrix ops.
//
// Fragment layouts per ISA 7.12.2 (16-bit, wave32):
//   A 16x32: lane l -> row m=l%16; lanes 0-15 hold K {0..7,16..23},
//            lanes 16-31 hold K {8..15,24..31} (2 x 16B runs -> ds_load_b128).
//   B 32x16: lane l -> column n=l%16; lanes 0-15 K {0..7,16..23}, lanes 16-31
//            K {8..15,24..31} (same K order as A, from row n of W[OUTC,128]).
//   C/D 16x16: VGPR r -> row m=r+8*(l/16), col n=l%16.
// ---------------------------------------------------------------------------
template <int OUTC, bool RELU, int RTILES>
__global__ void __launch_bounds__((OUTC / 16) * 32) sage_gemm(
    const float* __restrict__ Aagg,   // [N,128] unscaled neighbor sums
    const float* __restrict__ inv,    // [N]     1/max(deg,1)
    const float* __restrict__ Aself,  // [N,128] root features
    const float* __restrict__ Wl,     // [OUTC,128] row-major
    const float* __restrict__ Wr,     // [OUTC,128] row-major
    const float* __restrict__ bias,   // [OUTC]
    float* __restrict__ out)          // [N,OUTC]
{
    constexpr int NCT = OUTC / 16;       // column tiles == waves per block
    constexpr int T   = NCT * 32;        // threads per block

    __shared__ __align__(16) __bf16 sAgg[16 * 128];
    __shared__ __align__(16) __bf16 sSelf[16 * 128];

    const int tid  = threadIdx.x;
    const int lane = tid & 31;
    const int ct   = tid >> 5;           // wave id == column tile
    const int lo   = lane & 15;
    const int hi   = lane >> 4;
    const int n    = ct * 16 + lo;       // this lane's output column

    // ---- Preload both weight matrices for column n into registers (bf16) ---
    const float* wlRow = Wl + (size_t)n * 128;
    const float* wrRow = Wr + (size_t)n * 128;
    v16bf Bl[4], Br[4];
#pragma unroll
    for (int t = 0; t < 4; t++) {
        float wl[16], wr[16];
#pragma unroll
        for (int q = 0; q < 4; q++) {
            *(float4*)(wl + 4 * q) = *(const float4*)(wlRow + 32 * t + hi * 8 + 4 * (q & 1) + 16 * (q >> 1));
            *(float4*)(wr + 4 * q) = *(const float4*)(wrRow + 32 * t + hi * 8 + 4 * (q & 1) + 16 * (q >> 1));
        }
#pragma unroll
        for (int i = 0; i < 16; i++) {
            Bl[t][i] = (__bf16)wl[i];
            Br[t][i] = (__bf16)wr[i];
        }
    }
    const float bv = bias[n];

    // ---- Walk RTILES row tiles, staging A through LDS --------------------
    const int rt0 = blockIdx.x * RTILES;
    for (int r = 0; r < RTILES; r++) {
        const int rt = rt0 + r;

        // Cooperative stage: 16 rows x 128 cols, groups of 4 floats.
        // 512 groups per matrix; packed 4xbf16 -> single ds_store_b64 each.
        __syncthreads();   // previous iteration's reads done before overwrite
        union Pack { __bf16 b[4]; uint2 u; };
        for (int g = tid; g < 512; g += T) {
            const int m = g >> 5;
            const int cofs = (g & 31) * 4;
            const size_t row = (size_t)(rt * 16 + m) * 128;
            const float sc = inv[rt * 16 + m];
            float4 va = *(const float4*)(Aagg + row + cofs);
            float4 vs = *(const float4*)(Aself + row + cofs);
            Pack pa, ps;
            pa.b[0] = (__bf16)(va.x * sc); pa.b[1] = (__bf16)(va.y * sc);
            pa.b[2] = (__bf16)(va.z * sc); pa.b[3] = (__bf16)(va.w * sc);
            ps.b[0] = (__bf16)vs.x; ps.b[1] = (__bf16)vs.y;
            ps.b[2] = (__bf16)vs.z; ps.b[3] = (__bf16)vs.w;
            *(uint2*)(sAgg + m * 128 + cofs)  = pa.u;
            *(uint2*)(sSelf + m * 128 + cofs) = ps.u;
        }
        __syncthreads();

        // Issue ALL A-fragment loads first (16x ds_load_b128), then the 8
        // WMMAs — lets the scheduler overlap DS latency with matrix ops.
        union Frag { v16bf v; struct { float4 lo; float4 hi; } q; };
        Frag fa[4], fs[4];
        const __bf16* aBase = sAgg  + lo * 128 + hi * 8;
        const __bf16* sBase = sSelf + lo * 128 + hi * 8;
#pragma unroll
        for (int t = 0; t < 4; t++) {
            fa[t].q.lo = *(const float4*)(aBase + 32 * t);
            fa[t].q.hi = *(const float4*)(aBase + 32 * t + 16);
            fs[t].q.lo = *(const float4*)(sBase + 32 * t);
            fs[t].q.hi = *(const float4*)(sBase + 32 * t + 16);
        }
        v8f c = {};
#pragma unroll
        for (int t = 0; t < 4; t++) {
            c = __builtin_amdgcn_wmma_f32_16x16x32_bf16(false, fa[t].v, false, Bl[t],
                                                        (short)0, c, false, false);
            c = __builtin_amdgcn_wmma_f32_16x16x32_bf16(false, fs[t].v, false, Br[t],
                                                        (short)0, c, false, false);
        }

        // Bias (+ReLU); one base address, 8 stores at immediate offsets.
        float* po = out + (size_t)(rt * 16 + 8 * hi) * OUTC + n;
#pragma unroll
        for (int q = 0; q < 8; q++) {
            float v = c[q] + bv;
            if (RELU) v = fmaxf(v, 0.0f);
            po[q * OUTC] = v;
        }
    }
}

// ---------------------------------------------------------------------------
// Decode: out[e] = dot(z[src_e], z[dst_e]) over 64 dims. z rows are 256B
// aligned; z (26MB) is L2-resident.
// ---------------------------------------------------------------------------
__global__ void decode_dot(const float* __restrict__ z,
                           const int* __restrict__ ls,
                           const int* __restrict__ ld,
                           float* __restrict__ out, int L) {
    int e = blockIdx.x * blockDim.x + threadIdx.x;
    if (e >= L) return;
    const float4* a = (const float4*)(z + (size_t)ls[e] * 64);
    const float4* b = (const float4*)(z + (size_t)ld[e] * 64);
    float acc = 0.0f;
#pragma unroll
    for (int i = 0; i < 16; i++) {
        float4 u = a[i], v = b[i];
        acc += u.x * v.x + u.y * v.y + u.z * v.z + u.w * v.w;
    }
    out[e] = acc;
}

// ---------------------------------------------------------------------------
extern "C" void kernel_launch(void* const* d_in, const int* in_sizes, int n_in,
                              void* d_out, int out_size, void* d_ws, size_t ws_size,
                              hipStream_t stream) {
    (void)in_sizes; (void)n_in; (void)out_size; (void)ws_size;
    const float* x   = (const float*)d_in[0];
    const int*   ei  = (const int*)d_in[1];   // [2,E] flat (harness: integer -> int32)
    const int*   eli = (const int*)d_in[2];   // [2,L] flat
    const float* W1l = (const float*)d_in[3];
    const float* b1  = (const float*)d_in[4];
    const float* W1r = (const float*)d_in[5];
    const float* W2l = (const float*)d_in[6];
    const float* b2  = (const float*)d_in[7];
    const float* W2r = (const float*)d_in[8];
    float* outp = (float*)d_out;

    // Workspace carve-up (~128.5 MB total), 256B aligned slices.
    char*  ws  = (char*)d_ws;
    size_t off = 0;
    auto carve = [&](size_t bytes) -> float* {
        float* p = (float*)(ws + off);
        off = (off + bytes + 255) & ~(size_t)255;
        return p;
    };
    float* agg = carve((size_t)N_NODES * 128 * sizeof(float)); // 51.2 MB (reused both layers)
    float* cnt = carve((size_t)N_NODES * sizeof(float));       //  0.4 MB (deg -> 1/deg)
    float* h   = carve((size_t)N_NODES * 128 * sizeof(float)); // 51.2 MB
    float* z   = carve((size_t)N_NODES * 64 * sizeof(float));  // 25.6 MB

    const int* src = ei;
    const int* dst = ei + N_EDGES;
    const int* ls  = eli;
    const int* ld  = eli + N_LABEL;

    constexpr int RT = 5;                    // row tiles per block (6250 = 1250*5)
    constexpr int NBLK = (N_NODES / 16) / RT;

    // Degree + reciprocal (shared by both layers; recomputed every call for
    // graph-replay determinism).
    hipMemsetAsync(cnt, 0, (size_t)N_NODES * sizeof(float), stream);
    hipMemsetAsync(agg, 0, (size_t)N_NODES * 128 * sizeof(float), stream);
    count_edges<<<(N_EDGES + 255) / 256, 256, 0, stream>>>(dst, cnt, N_EDGES);
    inv_counts<<<(N_NODES + 255) / 256, 256, 0, stream>>>(cnt, N_NODES);

    // Layer 1: mean-aggregate x, fused dual-GEMM + bias + ReLU -> h
    scatter_add_feats<<<(N_EDGES * 32 + 255) / 256, 256, 0, stream>>>(x, src, dst, agg, N_EDGES);
    sage_gemm<128, true, RT><<<NBLK, 256, 0, stream>>>(agg, cnt, x, W1l, W1r, b1, h);

    // Layer 2: mean-aggregate h, fused dual-GEMM + bias -> z
    hipMemsetAsync(agg, 0, (size_t)N_NODES * 128 * sizeof(float), stream);
    scatter_add_feats<<<(N_EDGES * 32 + 255) / 256, 256, 0, stream>>>(h, src, dst, agg, N_EDGES);
    sage_gemm<64, false, RT><<<NBLK, 128, 0, stream>>>(agg, cnt, h, W2l, W2r, b2, z);

    // Decode: endpoint dot products
    decode_dot<<<(N_LABEL + 255) / 256, 256, 0, stream>>>(z, ls, ld, outp, N_LABEL);
}
